// PointCloudTo2D_54365696033547
// MI455X (gfx1250) — matrix operand order repaired and checked by reading, not compiled, not run on hardware
//
#include <hip/hip_runtime.h>

typedef unsigned short u16;
typedef __attribute__((ext_vector_type(16))) __bf16 v16bf;
typedef __attribute__((ext_vector_type(8)))  float  v8f;
typedef __attribute__((ext_vector_type(4)))  unsigned int v4u;
typedef __attribute__((ext_vector_type(8)))  int v8i;
typedef __attribute__((ext_vector_type(4)))  int v4i;

struct __align__(16) U128 { unsigned d[4]; };
union AFrag { v16bf v; U128 q[2]; };

__device__ __forceinline__ float bf2f(u16 s) {
  unsigned u = ((unsigned)s) << 16; float f; __builtin_memcpy(&f, &u, 4); return f;
}
__device__ __forceinline__ u16 f2bf(float f) {
  unsigned u; __builtin_memcpy(&u, &f, 4);
  u = u + 0x7FFFu + ((u >> 16) & 1u);
  return (u16)(u >> 16);
}
__device__ __forceinline__ float mishf(float x) {
  float sp = (x > 15.f) ? x : log1pf(expf(x));
  return x * tanhf(sp);
}

// padded activation tensor: [n][y+2][x+2][c], 132x132x64 bf16 per image
#define PH 132
#define PSTR ((size_t)PH * 64)            // row stride (elements)
#define PIMG ((size_t)PH * PH * 64)       // image stride (elements)

// ---------------------------------------------------------------- zero fill
__global__ void k_zero(float* out, long n) {
  long i = (long)blockIdx.x * blockDim.x + threadIdx.x;
  long stride = (long)gridDim.x * blockDim.x;
  for (; i < n; i += stride) out[i] = 0.f;
}

// --------------------------- pre-swizzle W2 into WMMA B-fragment bf16 layout
// wsw[ochalf][frag][lane][slot]; frag = ((tap*2 + kstep)*2 + octile)
__global__ void k_wprep(const float* __restrict__ W2, u16* __restrict__ wsw) {
  unsigned i = blockIdx.x * blockDim.x + threadIdx.x;   // 0 .. 73727
  int ochalf = i >> 15 >> 1;                            // i / 65536
  int r      = i & 36863;
  int f      = r >> 9;
  int lane   = (r >> 4) & 31;
  int slot   = r & 15;
  int octile = f & 1;
  int kstep  = (f >> 1) & 1;
  int tap    = f >> 2;
  int oc = ochalf * 32 + octile * 16 + (lane & 15);     // B col = lane%16
  int ic = kstep * 32 + (lane >> 4) * 8 + (slot & 7) + (slot >> 3) * 16;        // B row = K
  wsw[i] = f2bf(W2[((size_t)oc * 64 + ic) * 9 + tap]);  // OIHW, tap = ky*3+kx
}

// ------------------------------------------- conv1 (3->64) + mish + maxpool3s2
// rgb: NCHW f32 256x256.  out t2p: halo-padded NHWC bf16 (interior 128x128x64).
__global__ void k_conv1(const float* __restrict__ rgb, const float* __restrict__ W1,
                        u16* __restrict__ t2p) {
  __shared__ float sW[64 * 27];                       // OIHW flattened: oc*27 + ic*9 + tap
  for (int i = threadIdx.x; i < 64 * 27; i += blockDim.x) sW[i] = W1[i];
  __syncthreads();

  unsigned gid = blockIdx.x * blockDim.x + threadIdx.x;
  int g = gid & 3;
  int x = (gid >> 2) & 127;
  int y = (gid >> 9) & 127;
  int n = gid >> 16;

  float mx[16];
#pragma unroll
  for (int c = 0; c < 16; ++c) mx[c] = -3.0e38f;

  const float* rb = rgb + (size_t)n * 3 * 256 * 256;
  for (int wy = 0; wy < 3; ++wy) {
    int yc = 2 * y - 1 + wy;
    if (yc < 0 || yc >= 256) continue;
    for (int wx = 0; wx < 3; ++wx) {
      int xc = 2 * x - 1 + wx;
      if (xc < 0 || xc >= 256) continue;
      float acc[16];
#pragma unroll
      for (int c = 0; c < 16; ++c) acc[c] = 0.f;
      for (int ic = 0; ic < 3; ++ic) {
#pragma unroll
        for (int t = 0; t < 9; ++t) {
          int yi = yc + 2 * ((t / 3) - 1);            // dilation 2, pad 2
          int xi = xc + 2 * ((t % 3) - 1);
          float in = 0.f;
          if (yi >= 0 && yi < 256 && xi >= 0 && xi < 256)
            in = rb[((size_t)ic * 256 + yi) * 256 + xi];
          const float* wp = &sW[(g * 16) * 27 + ic * 9 + t];
#pragma unroll
          for (int c = 0; c < 16; ++c) acc[c] = fmaf(wp[c * 27], in, acc[c]);
        }
      }
#pragma unroll
      for (int c = 0; c < 16; ++c) mx[c] = fmaxf(mx[c], mishf(acc[c]));
    }
  }
  u16* op = t2p + (size_t)n * PIMG + (size_t)(y + 2) * PSTR + (size_t)(x + 2) * 64 + g * 16;
#pragma unroll
  for (int c = 0; c < 16; ++c) op[c] = f2bf(mx[c]);
}

// -------------------------------------------------- conv2 (64->64) via WMMA
// t2p: halo-padded NHWC bf16.  t3: NHWC bf16 128x128x64 (mish applied).
// Block = (y row, image n, oc-half). 8 waves, each a 16-pixel tile of the row.
// Weights: pre-swizzled bf16 fragments DMA'd into LDS by the Tensor Data Mover.
__global__ void k_conv2(const u16* __restrict__ t2p, const u16* __restrict__ wsw,
                        u16* __restrict__ t3) {
  __shared__ __align__(16) u16 sB[36 * 512];          // 36 KB of B fragments
  int ochalf = blockIdx.z;

  if (threadIdx.x < 32) {
    // TDM: copy 36 KB (4608 x 8B, 1 row) from wsw[ochalf] into sB
    unsigned long long ga = (unsigned long long)(uintptr_t)(wsw + (size_t)ochalf * 36864);
    unsigned ldsa = (unsigned)(uintptr_t)&sB[0];      // flat->LDS: low 32 bits
    v4u g0 = { 1u,                                    // count=1 (valid user D#)
               ldsa,                                  // lds_addr
               (unsigned)(ga & 0xFFFFFFFFu),          // global_addr[31:0]
               (unsigned)((ga >> 32) & 0x01FFFFFFu) | 0x80000000u }; // [56:32] | type=2
    v8i g1 = { 0x00030000,                            // data_size=8B, mask=0, flags=0
               0x12000000,                            // tensor_dim0 lo16 (4608) << 16
               0x00010000,                            // tensor_dim1 = 1
               0x12000000,                            // tile_dim0 = 4608
               0x00000001,                            // tile_dim1 = 1
               4608, 0, 0 };                          // tensor_dim0_stride = 4608
    v4i gz = { 0, 0, 0, 0 };
#if __clang_major__ >= 23
    v8i gz8 = { 0, 0, 0, 0, 0, 0, 0, 0 };
    __builtin_amdgcn_tensor_load_to_lds(g0, g1, gz, gz, gz8, 0);
#else
    __builtin_amdgcn_tensor_load_to_lds(g0, g1, gz, gz, 0);
#endif
    __builtin_amdgcn_s_wait_tensorcnt(0);
  }
  __syncthreads();

  int y    = blockIdx.x;
  int n    = blockIdx.y;
  int wave = threadIdx.x >> 5;
  int lane = threadIdx.x & 31;
  int half = lane >> 4;
  int x0   = wave * 16;
  int xm   = x0 + (lane & 15);                        // A row (pixel) for this lane

  v8f acc0 = {}, acc1 = {};
  const u16* tbase = t2p + (size_t)n * PIMG;

#pragma unroll
  for (int tap = 0; tap < 9; ++tap) {
    int dy = 2 * ((tap / 3) - 1);
    int dx = 2 * ((tap % 3) - 1);
    // halo offset: padded coords are (y+dy+2, xm+dx+2) — always in-bounds
    const u16* row = tbase + (size_t)(y + dy + 2) * PSTR + (size_t)(xm + dx + 2) * 64;
#pragma unroll
    for (int kstep = 0; kstep < 2; ++kstep) {
      AFrag a;
      int icb = kstep * 32 + half * 8;                // A lane halves: K {h8..h8+7, 16+h8..}
      a.q[0] = *(const U128*)(row + icb);
      a.q[1] = *(const U128*)(row + icb + 16);
#pragma unroll
      for (int o = 0; o < 2; ++o) {
        AFrag b;
        const U128* wp = (const U128*)(sB + ((tap * 2 + kstep) * 2 + o) * 512 + lane * 16);
        b.q[0] = wp[0];
        b.q[1] = wp[1];
        if (o == 0)
          acc0 = __builtin_amdgcn_wmma_f32_16x16x32_bf16(false, a.v, false, b.v,
                                                         (short)0, acc0, false, false);
        else
          acc1 = __builtin_amdgcn_wmma_f32_16x16x32_bf16(false, a.v, false, b.v,
                                                         (short)0, acc1, false, false);
      }
    }
  }

  // Epilogue: D layout — lane holds col N=lane%16, VGPR r holds row M=r+8*half.
  u16* obase = t3 + (((size_t)n * 128 + y) * 128) * 64;
#pragma unroll
  for (int r = 0; r < 8; ++r) {
    int xp = x0 + r + 8 * half;
    int oc = ochalf * 32 + (lane & 15);
    obase[(size_t)xp * 64 + oc]      = f2bf(mishf(acc0[r]));
    obase[(size_t)xp * 64 + oc + 16] = f2bf(mishf(acc1[r]));
  }
}

// ------------------------------------------------------- maxpool3s2 128->64
__global__ void k_pool2(const u16* __restrict__ t3, u16* __restrict__ t4) {
  unsigned gid = blockIdx.x * blockDim.x + threadIdx.x;
  int cg = gid & 7;
  int x  = (gid >> 3) & 63;
  int y  = (gid >> 9) & 63;
  int n  = gid >> 15;
  float mx[8];
#pragma unroll
  for (int c = 0; c < 8; ++c) mx[c] = -3.0e38f;
  const u16* base = t3 + (size_t)n * 128 * 128 * 64;
  for (int wy = 0; wy < 3; ++wy) {
    int yi = 2 * y - 1 + wy;
    if (yi < 0 || yi >= 128) continue;
    for (int wx = 0; wx < 3; ++wx) {
      int xi = 2 * x - 1 + wx;
      if (xi < 0 || xi >= 128) continue;
      const u16* p = base + ((size_t)yi * 128 + xi) * 64 + cg * 8;
#pragma unroll
      for (int c = 0; c < 8; ++c) mx[c] = fmaxf(mx[c], bf2f(p[c]));
    }
  }
  u16* op = t4 + (((size_t)n * 64 + y) * 64 + x) * 64 + cg * 8;
#pragma unroll
  for (int c = 0; c < 8; ++c) op[c] = f2bf(mx[c]);
}

// ------------------------------------------------------------- voxel scatter
__global__ void k_scatter(const float* __restrict__ rgb, const float* __restrict__ sxyz,
                          const int* __restrict__ index, const u16* __restrict__ t4,
                          float* __restrict__ out) {
  const long CNT_BASE = 16777216L;                    // 8*64*8*64*64
  const long RGB_BASE = 17039360L;                    // CNT_BASE + 8*8*64*64
  unsigned gid = blockIdx.x * blockDim.x + threadIdx.x;
  int t = gid & 63;
  unsigned p = gid >> 6;
  int xs = p & 63;
  int ys = (p >> 6) & 63;
  int n  = p >> 12;

  const float* xb = sxyz + (size_t)n * 3 * 256 * 256;
  int yi = 4 * ys + 2, xi = 4 * xs + 2;               // surf_xyz[:, :, 2::4, 2::4]
  float hx = xb[((size_t)0 * 256 + yi) * 256 + xi];
  float wx = xb[((size_t)1 * 256 + yi) * 256 + xi];
  float vx = xb[((size_t)2 * 256 + yi) * 256 + xi];
  int hs = (int)((hx + 5.0f) * (64.0f / 10.0f));      // trunc-toward-zero == astype(int32)
  int ws = (int)((wx + 5.0f) * (64.0f / 10.0f));
  int vs = (int)(vx * (8.0f / 3.0f));
  bool valid = (hs >= 0) & (hs < 64) & (ws >= 0) & (ws < 64) & (vs >= 0) & (vs < 8);
  if (!valid) return;

  int b = index[n];
  long cell = ((long)vs * 64 + hs) * 64 + ws;

  float f = bf2f(t4[(((size_t)n * 64 + ys) * 64 + xs) * 64 + t]);
  unsafeAtomicAdd(out + (long)(b * 64 + t) * (8 * 64 * 64) + cell, f);

  if (t < 3) {
    const float* rb = rgb + ((size_t)n * 3 + t) * 256 * 256;   // 4x4 mean pool
    float s = 0.f;
#pragma unroll
    for (int i = 0; i < 4; ++i)
#pragma unroll
      for (int j = 0; j < 4; ++j)
        s += rb[(size_t)(4 * ys + i) * 256 + (4 * xs + j)];
    s *= (1.0f / 16.0f);
    unsafeAtomicAdd(out + RGB_BASE + (long)(b * 3 + t) * (8 * 64 * 64) + cell, s);
  }
  if (t == 0)
    unsafeAtomicAdd(out + CNT_BASE + (long)b * (8 * 64 * 64) + cell, 1.0f);
}

// -----------------------------------------------------------------------------
extern "C" void kernel_launch(void* const* d_in, const int* in_sizes, int n_in,
                              void* d_out, int out_size, void* d_ws, size_t ws_size,
                              hipStream_t stream) {
  (void)n_in; (void)ws_size;
  const float* rgb   = (const float*)d_in[0];
  const float* sxyz  = (const float*)d_in[1];
  const float* W1    = (const float*)d_in[2];
  const float* W2    = (const float*)d_in[3];
  const int*   index = (const int*)d_in[4];
  int N = in_sizes[0] / (3 * 256 * 256);              // 64
  float* out = (float*)d_out;

  // workspace: t2p (halo 132x132x64 bf16), t3 (128^2x64 bf16), t4 (64^2x64 bf16),
  //            wsw (swizzled W2 fragments, 2 x 36 KB)
  u16* t2p = (u16*)d_ws;
  u16* t3  = t2p + (size_t)N * PIMG;
  u16* t4  = t3  + (size_t)N * 128 * 128 * 64;
  u16* wsw = t4  + (size_t)N * 64 * 64 * 64;

  k_zero<<<4096, 256, 0, stream>>>(out, (long)out_size);
  // zero padded activations (halo must be 0 every launch; ws is not re-poisoned)
  k_zero<<<8192, 256, 0, stream>>>((float*)t2p, (long)((size_t)N * PIMG / 2));
  k_wprep<<<288, 256, 0, stream>>>(W2, wsw);
  k_conv1<<<N * 256, 256, 0, stream>>>(rgb, W1, t2p);
  k_conv2<<<dim3(128, N, 2), 256, 0, stream>>>(t2p, wsw, t3);
  k_pool2<<<N * 128, 256, 0, stream>>>(t3, t4);
  k_scatter<<<N * 1024, 256, 0, stream>>>(rgb, sxyz, index, t4, out);
}
